// LowRankCayleyLinear_69630009803034
// MI455X (gfx1250) — matrix operand order repaired and checked by reading, not compiled, not run on hardware
//
#include <hip/hip_runtime.h>

typedef unsigned short u16;
typedef __attribute__((ext_vector_type(16))) __bf16 v16bf;
typedef __attribute__((ext_vector_type(8)))  float  v8f;
typedef __attribute__((ext_vector_type(4)))  int    i32x4;

#define NMAT 2048
#define LDSS 40    // LDS tile row stride (bf16 elems): 80 B, multiple of 16 B
#define LDSC 136   // LDS transpose-out row stride:     272 B, multiple of 16 B
#define TILE_ELEMS (128 * LDSS)          // 5120 elems per A or B tile
#define BUF_ELEMS  (2 * TILE_ELEMS)      // A+B per buffer

#if __has_builtin(__builtin_amdgcn_global_load_async_to_lds_b128) && \
    __has_builtin(__builtin_amdgcn_s_wait_asynccnt)
#define USE_ASYNC 1
#else
#define USE_ASYNC 0
#endif

#define AS1 __attribute__((address_space(1)))
#define AS3 __attribute__((address_space(3)))

__device__ __forceinline__ u16 f2bf(float f) {
  unsigned int u = __builtin_bit_cast(unsigned int, f);
  u += 0x7FFFu + ((u >> 16) & 1u);         // round-to-nearest-even
  return (u16)(u >> 16);
}

// copy 16 bf16 (32 B) global -> LDS
__device__ __forceinline__ void stage16(const u16* __restrict__ g, u16* l) {
#if USE_ASYNC
  AS1 i32x4* gp = (AS1 i32x4*)g;
  AS3 i32x4* lp = (AS3 i32x4*)l;
  __builtin_amdgcn_global_load_async_to_lds_b128(gp, lp, 0, 0);
  __builtin_amdgcn_global_load_async_to_lds_b128(gp + 1, lp + 1, 0, 0);
#else
  uint4 q0 = ((const uint4*)g)[0];
  uint4 q1 = ((const uint4*)g)[1];
  ((uint4*)l)[0] = q0;
  ((uint4*)l)[1] = q1;
#endif
}

__device__ __forceinline__ void wait_stage() {
#if USE_ASYNC
  __builtin_amdgcn_s_wait_asynccnt(0);
#endif
}

union Frag { v16bf v; uint4 q[2]; };

// ---------------------------------------------------------------------------
// A = U V^T - V U^T (skew).  Emit bf16:  Bm = I-A,  IpA = I+A,
// X1 = I+A (first NS iter folded),  X1T = (I+A)^T = I-A.
// ---------------------------------------------------------------------------
__global__ void build_mats(const float* __restrict__ U, const float* __restrict__ V,
                           u16* __restrict__ Bm, u16* __restrict__ IpA,
                           u16* __restrict__ X, u16* __restrict__ XT) {
  int j = blockIdx.x * 16 + threadIdx.x;
  int i = blockIdx.y * 16 + threadIdx.y;
  const float* ui = U + (size_t)i * 64;
  const float* vi = V + (size_t)i * 64;
  const float* uj = U + (size_t)j * 64;
  const float* vj = V + (size_t)j * 64;
  float a = 0.f;
#pragma unroll 8
  for (int k = 0; k < 64; ++k) a += ui[k] * vj[k] - vi[k] * uj[k];
  float d = (i == j) ? 1.0f : 0.0f;
  size_t idx = (size_t)i * NMAT + j;
  u16 m = f2bf(d - a);
  u16 p = f2bf(d + a);
  Bm[idx]  = m;
  IpA[idx] = p;
  X[idx]   = p;
  XT[idx]  = m;   // X1^T = I - A
}

// ---------------------------------------------------------------------------
// acc = P @ Q with Q supplied TRANSPOSED row-major (QT[n][k] = Q[k][n]).
// mode==1: acc := 2I - acc.  Outputs (each nullable):
//   C  : row-major;  CT : transposed via LDS re-tile -> global_store_b128.
// 256 threads / 8 waves, 128x128 block tile, wave owns 32x64 (2x4 WMMA tiles).
// Double-buffered LDS + async global->LDS staging (ASYNCcnt tracked).
// ---------------------------------------------------------------------------
__global__ __launch_bounds__(256) void gemm_nsq(const u16* __restrict__ P,
                                                const u16* __restrict__ QT,
                                                u16* __restrict__ C,
                                                u16* __restrict__ CT, int mode) {
  __shared__ __align__(16) u16 smem[2 * BUF_ELEMS];    // 40960 B
  u16 (*lCT)[LDSC] = (u16(*)[LDSC])smem;               // epilogue reuse (34816 B)

  const int t    = threadIdx.x;
  const int lane = t & 31;
  const int wave = t >> 5;
  const int half = lane >> 4;
  const int l15  = lane & 15;
  const int wm   = (wave >> 1) * 32;
  const int wn   = (wave & 1) * 64;
  const int bm   = blockIdx.y * 128;
  const int bn   = blockIdx.x * 128;

  const int srow = t >> 1;                 // staging row (0..127)
  const int sseg = (t & 1) * 16;           // staging K segment (0 or 16)
  const size_t gA = (size_t)(bm + srow) * NMAT + sseg;
  const size_t gB = (size_t)(bn + srow) * NMAT + sseg;
  const int   lof = srow * LDSS + sseg;

  v8f zero = {};
  v8f acc[2][4];
#pragma unroll
  for (int i = 0; i < 2; ++i)
#pragma unroll
    for (int j = 0; j < 4; ++j) acc[i][j] = zero;

  // prologue: stage tile 0 into buffer 0
  stage16(P  + gA, smem + lof);
  stage16(QT + gB, smem + TILE_ELEMS + lof);

  for (int k0 = 0; k0 < NMAT; k0 += 32) {
    const int cur = (k0 >> 5) & 1;
    u16* bufA = smem + cur * BUF_ELEMS;
    u16* bufB = bufA + TILE_ELEMS;

    wait_stage();          // tile k0's async copies have landed
    __syncthreads();       // ...visible to everyone; everyone done reading buf^1

    if (k0 + 32 < NMAT) {  // stage next tile into the other buffer
      u16* nA = smem + (cur ^ 1) * BUF_ELEMS;
      stage16(P  + gA + k0 + 32, nA + lof);
      stage16(QT + gB + k0 + 32, nA + TILE_ELEMS + lof);
    }

    Frag a[2], b[4];
#pragma unroll
    for (int i = 0; i < 2; ++i) { // A lane layout: K chunks {half*8, half*8+16}
      const u16* r = bufA + (wm + i * 16 + l15) * LDSS + half * 8;
      a[i].q[0] = *(const uint4*)r;
      a[i].q[1] = *(const uint4*)(r + 16);
    }
#pragma unroll
    for (int j = 0; j < 4; ++j) { // B lane layout: K chunks {half*8, half*8+16}
      const u16* r = bufB + (wn + j * 16 + l15) * LDSS + half * 8;
      b[j].q[0] = *(const uint4*)r;
      b[j].q[1] = *(const uint4*)(r + 16);
    }
#pragma unroll
    for (int i = 0; i < 2; ++i)
#pragma unroll
      for (int j = 0; j < 4; ++j)
        acc[i][j] = __builtin_amdgcn_wmma_f32_16x16x32_bf16(
            false, a[i].v, false, b[j].v, (short)0, acc[i][j], false, false);
  }

  // ---- epilogue -----------------------------------------------------------
  if (mode) { // acc := 2I - acc
#pragma unroll
    for (int i = 0; i < 2; ++i)
#pragma unroll
      for (int j = 0; j < 4; ++j)
#pragma unroll
        for (int r = 0; r < 8; ++r) {
          int row = bm + wm + i * 16 + half * 8 + r;
          int col = bn + wn + j * 16 + l15;
          acc[i][j][r] = ((row == col) ? 2.0f : 0.0f) - acc[i][j][r];
        }
  }

  if (C != nullptr) {
#pragma unroll
    for (int i = 0; i < 2; ++i)
#pragma unroll
      for (int j = 0; j < 4; ++j)
#pragma unroll
        for (int r = 0; r < 8; ++r) {
          int row = bm + wm + i * 16 + half * 8 + r;
          int col = bn + wn + j * 16 + l15;
          C[(size_t)row * NMAT + col] = f2bf(acc[i][j][r]);
        }
  }

  if (CT != nullptr) { // transpose 128x128 block through LDS, store coalesced
    __syncthreads();   // everyone done with the tile buffers
#pragma unroll
    for (int i = 0; i < 2; ++i)
#pragma unroll
      for (int j = 0; j < 4; ++j)
#pragma unroll
        for (int r = 0; r < 8; ++r) {
          int lr = wm + i * 16 + half * 8 + r;
          int lc = wn + j * 16 + l15;
          lCT[lc][lr] = f2bf(acc[i][j][r]);
        }
    __syncthreads();
    int n = t >> 1, seg = (t & 1) * 64;          // 64 elems = 128 B per thread
    const uint4* s4 = (const uint4*)&lCT[n][seg];
    uint4* dst = (uint4*)(CT + (size_t)(bn + n) * NMAT + bm + seg);
#pragma unroll
    for (int e = 0; e < 8; ++e) dst[e] = s4[e];
  }
}

// ---------------------------------------------------------------------------
// out[m,n] = sum_k inp[m,k] * W[n,k]  (inp f32 16384x2048, W bf16 2048x2048).
// A staged with in-register f32->bf16 convert (sync); B staged async.
// ---------------------------------------------------------------------------
__global__ __launch_bounds__(256) void gemm_out(const float* __restrict__ inp,
                                                const u16* __restrict__ W,
                                                float* __restrict__ out) {
  __shared__ __align__(16) u16 smem[2 * BUF_ELEMS];
  const int t    = threadIdx.x;
  const int lane = t & 31;
  const int wave = t >> 5;
  const int half = lane >> 4;
  const int l15  = lane & 15;
  const int wm   = (wave >> 1) * 32;
  const int wn   = (wave & 1) * 64;
  const int bm   = blockIdx.y * 128;   // M = 16384
  const int bn   = blockIdx.x * 128;   // N = 2048

  const int srow = t >> 1;
  const int sseg = (t & 1) * 16;
  const size_t gA = (size_t)(bm + srow) * NMAT + sseg;
  const size_t gB = (size_t)(bn + srow) * NMAT + sseg;
  const int   lof = srow * LDSS + sseg;

  v8f zero = {};
  v8f acc[2][4];
#pragma unroll
  for (int i = 0; i < 2; ++i)
#pragma unroll
    for (int j = 0; j < 4; ++j) acc[i][j] = zero;

  auto stageA = [&](u16* buf, int kk) {   // f32 -> bf16 conversion staging
    const float* src = inp + gA + kk;
    u16 tmp[16];
#pragma unroll
    for (int e = 0; e < 16; e += 4) {
      float4 f = *(const float4*)(src + e);
      tmp[e]     = f2bf(f.x);
      tmp[e + 1] = f2bf(f.y);
      tmp[e + 2] = f2bf(f.z);
      tmp[e + 3] = f2bf(f.w);
    }
    *(uint4*)(buf + lof)     = *(uint4*)&tmp[0];
    *(uint4*)(buf + lof + 8) = *(uint4*)&tmp[8];
  };

  // prologue: tile 0 into buffer 0
  stageA(smem, 0);
  stage16(W + gB, smem + TILE_ELEMS + lof);

  for (int k0 = 0; k0 < NMAT; k0 += 32) {
    const int cur = (k0 >> 5) & 1;
    u16* bufA = smem + cur * BUF_ELEMS;
    u16* bufB = bufA + TILE_ELEMS;

    wait_stage();
    __syncthreads();

    if (k0 + 32 < NMAT) {
      u16* nA = smem + (cur ^ 1) * BUF_ELEMS;
      stageA(nA, k0 + 32);
      stage16(W + gB + k0 + 32, nA + TILE_ELEMS + lof);
    }

    Frag a[2], b[4];
#pragma unroll
    for (int i = 0; i < 2; ++i) {
      const u16* r = bufA + (wm + i * 16 + l15) * LDSS + half * 8;
      a[i].q[0] = *(const uint4*)r;
      a[i].q[1] = *(const uint4*)(r + 16);
    }
#pragma unroll
    for (int j = 0; j < 4; ++j) {
      const u16* r = bufB + (wn + j * 16 + l15) * LDSS + half * 8;
      b[j].q[0] = *(const uint4*)r;
      b[j].q[1] = *(const uint4*)(r + 16);
    }
#pragma unroll
    for (int i = 0; i < 2; ++i)
#pragma unroll
      for (int j = 0; j < 4; ++j)
        acc[i][j] = __builtin_amdgcn_wmma_f32_16x16x32_bf16(
            false, a[i].v, false, b[j].v, (short)0, acc[i][j], false, false);
  }

#pragma unroll
  for (int i = 0; i < 2; ++i)
#pragma unroll
    for (int j = 0; j < 4; ++j)
#pragma unroll
      for (int r = 0; r < 8; ++r) {
        int row = bm + wm + i * 16 + half * 8 + r;
        int col = bn + wn + j * 16 + l15;
        out[(size_t)row * NMAT + col] = acc[i][j][r];
      }
}

// ---------------------------------------------------------------------------
extern "C" void kernel_launch(void* const* d_in, const int* in_sizes, int n_in,
                              void* d_out, int out_size, void* d_ws, size_t ws_size,
                              hipStream_t stream) {
  const float* inp = (const float*)d_in[0];   // [8,2048,2048]
  const float* U   = (const float*)d_in[1];   // [2048,64]
  const float* V   = (const float*)d_in[2];   // [2048,64]
  float* out = (float*)d_out;                 // [8,2048,2048]

  const size_t MAT = (size_t)NMAT * NMAT;     // elems per bf16 matrix (8 MiB)
  u16* Bm  = (u16*)d_ws;                      // B = I - A
  u16* IpA = Bm  + MAT;                       // I + A
  u16* Xa  = IpA + MAT;                       // X ping
  u16* XaT = Xa  + MAT;                       // X ping, transposed
  u16* Xb  = XaT + MAT;                       // X pong
  u16* XbT = Xb  + MAT;                       // X pong, transposed
  u16* TT  = XbT + MAT;                       // (2I - B@X)^T
  // total 56 MiB of workspace

  build_mats<<<dim3(128, 128), dim3(16, 16), 0, stream>>>(U, V, Bm, IpA, Xa, XaT);

  dim3 g(16, 16), blk(256);
  u16* X = Xa;  u16* XT = XaT;                // X1 = I + A
  u16* Xo = Xb; u16* XoT = XbT;
  for (int it = 0; it < 3; ++it) {            // remaining 3 of 4 NS iterations
    // T = 2I - B@X : only T^T is ever consumed
    gemm_nsq<<<g, blk, 0, stream>>>(Bm, XT, nullptr, TT, 1);
    // X' = X@T : consumed both as P (normal) and as Q (transposed)
    gemm_nsq<<<g, blk, 0, stream>>>(X, TT, Xo, XoT, 0);
    u16* s;
    s = X;  X  = Xo;  Xo  = s;
    s = XT; XT = XoT; XoT = s;
  }
  u16* Wm = Xo;                               // free ping-pong buffer
  // W = (I+A)@X : only normal layout needed downstream
  gemm_nsq<<<g, blk, 0, stream>>>(IpA, XT, Wm, nullptr, 0);

  gemm_out<<<dim3(16, 128), blk, 0, stream>>>(inp, Wm, out);
}
